// PMA_481036337712
// MI455X (gfx1250) — matrix operand order repaired and checked by reading, not compiled, not run on hardware
//
#include <hip/hip_runtime.h>
#include <hip/hip_bf16.h>

typedef __attribute__((ext_vector_type(16))) _Float16 v16h;
typedef __attribute__((ext_vector_type(8)))  float    v8f;

#define B_ 64
#define S_ 512
#define D_ 256
#define H_ 4
static constexpr float EPS_   = 1e-6f;
static constexpr float SCALE_ = 0.0625f;   // 1/sqrt(256)

__device__ __forceinline__ v8f wmma_f16(v16h a, v16h b, v8f c) {
  // (neg_a, A, neg_b, B, c_mod, C, reuse_a, reuse_b)
  return __builtin_amdgcn_wmma_f32_16x16x32_f16(false, a, false, b, (short)0, c,
                                                false, false);
}

// K-offset within a 32-wide K-tile for A fragments (16-bit, 16x32):
// lane<16: halves 0..15 -> K 0..15; lane>=16: K 16..31.
__device__ __forceinline__ int a_koff(int j, int hi8) {
  return j + 2 * hi8;
}

// ---------------------------------------------------------------------------
// K1a: q[h,:] = seeds @ Wq[:,h,:] + bq[h]     (4 blocks, one per head)
// ---------------------------------------------------------------------------
__global__ __launch_bounds__(256) void pma_q_kernel(
    const float* __restrict__ seeds, const float* __restrict__ Wq,
    const float* __restrict__ bq, float* __restrict__ q) {
  const int h = blockIdx.x, e = threadIdx.x;
  float acc = bq[h * D_ + e];
#pragma unroll 4
  for (int d = 0; d < D_; ++d)
    acc += seeds[d] * Wq[(size_t)d * (H_ * D_) + h * D_ + e];
  q[h * D_ + e] = acc;
}

// ---------------------------------------------------------------------------
// K1b: w_eff[h,d] = Wk[d,h,:]·q[h] ; bk_dot[h] = bk[h]·q[h]  (4 blocks)
// ---------------------------------------------------------------------------
__global__ __launch_bounds__(256) void pma_weff_kernel(
    const float* __restrict__ Wk, const float* __restrict__ bk,
    const float* __restrict__ q, float* __restrict__ weff,
    float* __restrict__ bkdot) {
  __shared__ float qL[D_];
  __shared__ float red[256];
  const int h = blockIdx.x, t = threadIdx.x;
  const float qv = q[h * D_ + t];
  qL[t]  = qv;
  red[t] = bk[h * D_ + t] * qv;
  __syncthreads();
  float acc = 0.f;
  const float* wr = Wk + (size_t)t * (H_ * D_) + h * D_;
#pragma unroll 4
  for (int e = 0; e < D_; ++e) acc += wr[e] * qL[e];
  weff[h * D_ + t] = acc;
  for (int off = 128; off > 0; off >>= 1) {
    __syncthreads();
    if (t < off) red[t] += red[t + off];
  }
  if (t == 0) bkdot[h] = red[0];
}

// ---------------------------------------------------------------------------
// K2 (one block per batch item): scores -> softmax -> xbar
// Phase 1 streams x from HBM (prefetched); phase 2 re-reads via L2 with
// LDS-staged f16 tiles for conflict-free WMMA B fragments.
// ---------------------------------------------------------------------------
__global__ __launch_bounds__(256) void pma_attn_kernel(
    const float* __restrict__ x, const int* __restrict__ mask,
    const float* __restrict__ weff, const float* __restrict__ bkdot,
    float* __restrict__ xbar) {
  __shared__ float    sc[S_ * H_];      // raw dot products (8 KB)
  __shared__ _Float16 p[S_ * H_];       // normalized attention (4 KB)
  __shared__ float    red[256];         // (1 KB)
  __shared__ _Float16 xtile[32 * D_];   // staged x chunk (16 KB)

  const int b    = blockIdx.x;
  const int tid  = threadIdx.x;
  const int wave = tid >> 5;
  const int lane = tid & 31;
  const int l    = lane & 15;
  const int hi8  = (lane >= 16) ? 8 : 0;
  const int hi16 = (lane >= 16) ? 16 : 0;
  const float* xb = x + (size_t)b * S_ * D_;

  // ---- Phase 1: sc[s,h] = x[b,s,:] · w_eff[h,:]  (GEMM [S,256]x[256,16pad])
  for (int st = wave; st < S_ / 16; st += 8) {
    if (st + 8 < S_ / 16)  // prefetch next row-tile for this wave (HBM pass)
      __builtin_prefetch(xb + (size_t)((st + 8) * 16 + l) * D_, 0, 0);
    v8f c = {};
#pragma unroll
    for (int kt = 0; kt < D_ / 32; ++kt) {
      const int k0 = kt * 32;
      v16h a;
      const float* arow = xb + (size_t)(st * 16 + l) * D_ + k0;
#pragma unroll
      for (int j = 0; j < 16; ++j) a[j] = (_Float16)arow[a_koff(j, hi8)];
      v16h bb;
      if (l < H_) {
        const float* wcol = weff + l * D_ + k0 + hi16;
#pragma unroll
        for (int j = 0; j < 16; ++j) bb[j] = (_Float16)wcol[j];
      } else {
#pragma unroll
        for (int j = 0; j < 16; ++j) bb[j] = (_Float16)0.f;
      }
      c = wmma_f16(a, bb, c);
    }
    if (l < H_) {
#pragma unroll
      for (int r = 0; r < 8; ++r) sc[(st * 16 + r + hi8) * H_ + l] = c[r];
    }
  }
  __syncthreads();

  // ---- Softmax over s per head (each thread owns s=tid and s=tid+256)
  const float add0 = mask[b * S_ + tid] ? 0.f : -1e9f;
  const float add1 = mask[b * S_ + tid + 256] ? 0.f : -1e9f;
  for (int h = 0; h < H_; ++h) {
    const float bkd = bkdot[h];
    float a0 = (sc[tid * H_ + h] + bkd) * SCALE_ + add0;
    float a1 = (sc[(tid + 256) * H_ + h] + bkd) * SCALE_ + add1;
    red[tid] = fmaxf(a0, a1);
    __syncthreads();
    for (int off = 128; off > 0; off >>= 1) {
      if (tid < off) red[tid] = fmaxf(red[tid], red[tid + off]);
      __syncthreads();
    }
    const float mx = red[0];
    __syncthreads();
    const float e0 = __expf(a0 - mx), e1 = __expf(a1 - mx);
    red[tid] = e0 + e1;
    __syncthreads();
    for (int off = 128; off > 0; off >>= 1) {
      if (tid < off) red[tid] += red[tid + off];
      __syncthreads();
    }
    const float inv = 1.f / red[0];
    __syncthreads();
    p[tid * H_ + h]         = (_Float16)(e0 * inv);
    p[(tid + 256) * H_ + h] = (_Float16)(e1 * inv);
  }

  // ---- Phase 2: xbar = pT (16pad x S) · x[b] (S x D), x staged through LDS.
  // Each wave owns two 16-wide column tiles; block walks K (s) in lockstep.
  const int nt0 = wave * 2, nt1 = nt0 + 1;
  v8f c0 = {}, c1 = {};
  for (int kt = 0; kt < S_ / 32; ++kt) {
    const int k0 = kt * 32;
    __syncthreads();  // previous chunk fully consumed / p ready on first iter
    {                 // coalesced f32->f16 staging of x[b, k0..k0+31, :]
      const int r  = tid >> 3;
      const int cc = (tid & 7) * 32;
      const float* src = xb + (size_t)(k0 + r) * D_ + cc;
      _Float16* dst = xtile + r * D_ + cc;
#pragma unroll
      for (int u = 0; u < 8; ++u) {
        const float4 f = *(const float4*)(src + u * 4);
        dst[u * 4 + 0] = (_Float16)f.x;
        dst[u * 4 + 1] = (_Float16)f.y;
        dst[u * 4 + 2] = (_Float16)f.z;
        dst[u * 4 + 3] = (_Float16)f.w;
      }
    }
    __syncthreads();
    v16h a;
#pragma unroll
    for (int j = 0; j < 16; ++j) {
      const int kk = k0 + a_koff(j, hi8);
      a[j] = (l < H_) ? p[kk * H_ + l] : (_Float16)0.f;
    }
    v16h b0, b1;
#pragma unroll
    for (int j = 0; j < 16; ++j) {
      b0[j] = xtile[(hi16 + j) * D_ + nt0 * 16 + l];
      b1[j] = xtile[(hi16 + j) * D_ + nt1 * 16 + l];
    }
    c0 = wmma_f16(a, b0, c0);
    c1 = wmma_f16(a, b1, c1);
  }
  if (lane < 16) {  // rows 0..3 hold the H=4 heads
#pragma unroll
    for (int r = 0; r < H_; ++r) {
      xbar[((size_t)b * H_ + r) * D_ + nt0 * 16 + lane] = c0[r];
      xbar[((size_t)b * H_ + r) * D_ + nt1 * 16 + lane] = c1[r];
    }
  }
}

// ---------------------------------------------------------------------------
// K3a: O[b,h,:] = xbar[b,h,:] @ Wv[:,h,:] + bv[h]   (per-head 64x256x256 GEMM)
// ---------------------------------------------------------------------------
__global__ __launch_bounds__(256) void pma_projv_kernel(
    const float* __restrict__ xbar, const float* __restrict__ Wv,
    const float* __restrict__ bv, float* __restrict__ O) {
  const int tid = threadIdx.x, wave = tid >> 5, lane = tid & 31, l = lane & 15;
  const int hi8 = (lane >= 16) ? 8 : 0, hi16 = (lane >= 16) ? 16 : 0;
  const int task = blockIdx.x * 8 + wave;  // 256 tasks: h(4) x mt(4) x nt(16)
  const int h = task >> 6, mt = (task >> 4) & 3, nt = task & 15;
  const int e0 = nt * 16;
  v8f c = {};
#pragma unroll
  for (int kt = 0; kt < D_ / 32; ++kt) {
    const int k0 = kt * 32;
    v16h a;
    const float* ar = xbar + ((size_t)(mt * 16 + l) * H_ + h) * D_ + k0;
#pragma unroll
    for (int j = 0; j < 16; ++j) a[j] = (_Float16)ar[a_koff(j, hi8)];
    v16h bb;
    const float* bc = Wv + (size_t)(k0 + hi16) * (H_ * D_) + h * D_ + e0 + l;
#pragma unroll
    for (int j = 0; j < 16; ++j) bb[j] = (_Float16)bc[(size_t)j * (H_ * D_)];
    c = wmma_f16(a, bb, c);
  }
  const float bias = bv[h * D_ + e0 + l];
#pragma unroll
  for (int r = 0; r < 8; ++r) {
    const int brow = mt * 16 + r + hi8;
    O[((size_t)brow * H_ + h) * D_ + e0 + l] = c[r] + bias;
  }
}

// ---------------------------------------------------------------------------
// K3b: out_pre[b,:] = O[b,:(H*D)] @ Wo[(H*D),D] + bo + seeds ([64,1024]x[1024,256])
// ---------------------------------------------------------------------------
__global__ __launch_bounds__(256) void pma_projo_kernel(
    const float* __restrict__ O, const float* __restrict__ Wo,
    const float* __restrict__ bo, const float* __restrict__ seeds,
    float* __restrict__ outpre) {
  const int tid = threadIdx.x, wave = tid >> 5, lane = tid & 31, l = lane & 15;
  const int hi8 = (lane >= 16) ? 8 : 0, hi16 = (lane >= 16) ? 16 : 0;
  const int task = blockIdx.x * 8 + wave;  // 64 tasks: mt(4) x nt(16)
  const int mt = task >> 4, nt = task & 15;
  const int d0 = nt * 16;
  v8f c = {};
#pragma unroll 8
  for (int kt = 0; kt < (H_ * D_) / 32; ++kt) {
    const int k0 = kt * 32;
    v16h a;
    const float* ar = O + (size_t)(mt * 16 + l) * (H_ * D_) + k0;
#pragma unroll
    for (int j = 0; j < 16; ++j) a[j] = (_Float16)ar[a_koff(j, hi8)];
    v16h bb;
    const float* bc = Wo + (size_t)(k0 + hi16) * D_ + d0 + l;
#pragma unroll
    for (int j = 0; j < 16; ++j) bb[j] = (_Float16)bc[(size_t)j * D_];
    c = wmma_f16(a, bb, c);
  }
  const float add = bo[d0 + l] + seeds[d0 + l];
#pragma unroll
  for (int r = 0; r < 8; ++r)
    outpre[(size_t)(mt * 16 + r + hi8) * D_ + d0 + l] = c[r] + add;
}

// ---------------------------------------------------------------------------
// K4: LayerNorm over D per batch row
// ---------------------------------------------------------------------------
__global__ __launch_bounds__(256) void pma_ln_kernel(
    const float* __restrict__ outpre, const float* __restrict__ gamma,
    const float* __restrict__ beta, float* __restrict__ out) {
  __shared__ float red[256];
  const int b = blockIdx.x, t = threadIdx.x;
  const float v = outpre[(size_t)b * D_ + t];
  red[t] = v;
  __syncthreads();
  for (int off = 128; off > 0; off >>= 1) {
    if (t < off) red[t] += red[t + off];
    __syncthreads();
  }
  const float mu = red[0] * (1.f / D_);
  __syncthreads();
  const float dv = v - mu;
  red[t] = dv * dv;
  __syncthreads();
  for (int off = 128; off > 0; off >>= 1) {
    if (t < off) red[t] += red[t + off];
    __syncthreads();
  }
  const float var = red[0] * (1.f / D_);
  out[(size_t)b * D_ + t] = dv * rsqrtf(var + EPS_) * gamma[t] + beta[t];
}

extern "C" void kernel_launch(void* const* d_in, const int* in_sizes, int n_in,
                              void* d_out, int out_size, void* d_ws,
                              size_t ws_size, hipStream_t stream) {
  (void)in_sizes; (void)n_in; (void)out_size; (void)ws_size;
  const float* x     = (const float*)d_in[0];
  const int*   mask  = (const int*)d_in[1];
  const float* seeds = (const float*)d_in[2];
  const float* Wq    = (const float*)d_in[3];
  const float* bq    = (const float*)d_in[4];
  const float* Wk    = (const float*)d_in[5];
  const float* bk    = (const float*)d_in[6];
  const float* Wv    = (const float*)d_in[7];
  const float* bv    = (const float*)d_in[8];
  const float* Wo    = (const float*)d_in[9];
  const float* bo    = (const float*)d_in[10];
  const float* gamma = (const float*)d_in[11];
  const float* beta  = (const float*)d_in[12];
  float* out = (float*)d_out;

  float* ws     = (float*)d_ws;
  float* q      = ws;                       // [H,D]   1024 f32
  float* weff   = ws + 1024;                // [H,D]   1024 f32
  float* bkdot  = ws + 2048;                // [H]
  float* xbar   = ws + 4096;                // [B,H,D] 65536 f32
  float* O      = ws + 4096 + 65536;        // [B,H,D] 65536 f32
  float* outpre = ws + 4096 + 2 * 65536;    // [B,D]   16384 f32

  pma_q_kernel<<<H_, 256, 0, stream>>>(seeds, Wq, bq, q);
  pma_weff_kernel<<<H_, 256, 0, stream>>>(Wk, bk, q, weff, bkdot);
  pma_attn_kernel<<<B_, 256, 0, stream>>>(x, mask, weff, bkdot, xbar);
  pma_projv_kernel<<<32, 256, 0, stream>>>(xbar, Wv, bv, O);
  pma_projo_kernel<<<8, 256, 0, stream>>>(O, Wo, bo, seeds, outpre);
  pma_ln_kernel<<<B_, 256, 0, stream>>>(outpre, gamma, beta, out);
}
